// MLP_v1_83064667505168
// MI455X (gfx1250) — matrix operand order repaired and checked by reading, not compiled, not run on hardware
//
#include <hip/hip_runtime.h>
#include <hip/hip_bf16.h>
#include <math.h>

// ---------------------------------------------------------------------------
// MI455X MLP: 4-layer GEMM chain via v_wmma_f32_16x16x32_bf16 (wave32).
// Compute-bound (687 GFLOP vs ~0.5 GB traffic) -> WMMA bf16 path.
// Round 3: fix async-builtin pointer types (int4 in AS1/AS3), keep
//          ASYNCcnt double-buffered staging + v_tanh_f32 epilogue.
// ---------------------------------------------------------------------------

typedef __attribute__((ext_vector_type(16))) __bf16          v16bf;
typedef __attribute__((ext_vector_type(8)))  float           v8f;
typedef __attribute__((ext_vector_type(8)))  unsigned short  u16x8;
typedef __attribute__((ext_vector_type(4)))  unsigned int    u32x4;
typedef __attribute__((ext_vector_type(4)))  int             i32x4;

union AFrag { v16bf v; u16x8 h[2]; };

#if defined(__gfx1250__) && __has_builtin(__builtin_amdgcn_global_load_async_to_lds_b128)
#define USE_ASYNC_LDS 1
typedef __attribute__((address_space(1))) i32x4 gas_i32x4;   // global (device) AS
typedef __attribute__((address_space(3))) i32x4 las_i32x4;   // LDS AS
#else
#define USE_ASYNC_LDS 0
#endif

#if defined(__gfx1250__) && __has_builtin(__builtin_amdgcn_tanhf)
#define DEV_TANH(x) __builtin_amdgcn_tanhf(x)
#else
#define DEV_TANH(x) tanhf(x)
#endif

static __device__ __forceinline__ void async_wait0() {
#if USE_ASYNC_LDS
#if __has_builtin(__builtin_amdgcn_s_wait_asynccnt)
    __builtin_amdgcn_s_wait_asynccnt(0);
#else
    asm volatile("s_wait_asynccnt 0" ::: "memory");
#endif
#endif
}

static __device__ __forceinline__ unsigned short f32_to_bf16_rne(float f) {
    unsigned int u = __builtin_bit_cast(unsigned int, f);
    u += 0x7fffu + ((u >> 16) & 1u);            // round-to-nearest-even
    return (unsigned short)(u >> 16);
}

// ---------------------------------------------------------------------------
// f32 -> bf16 conversion, 4-wide (inputs are multiples of 1024 elements)
// ---------------------------------------------------------------------------
__global__ void cvt_bf16_kernel(const float4* __restrict__ in,
                                uint2* __restrict__ out, int n4) {
    int i = blockIdx.x * blockDim.x + threadIdx.x;
    int stride = gridDim.x * blockDim.x;
    for (; i < n4; i += stride) {
        float4 f = in[i];
        uint2 r;
        r.x = ((unsigned int)f32_to_bf16_rne(f.y) << 16) | f32_to_bf16_rne(f.x);
        r.y = ((unsigned int)f32_to_bf16_rne(f.w) << 16) | f32_to_bf16_rne(f.z);
        out[i] = r;
    }
}

// ---------------------------------------------------------------------------
// per-row limits(): lim = pos ? relu(x0-0.5*x1)*relu(2*x1-x0)/(0.5*x1)^2 : 0
// ---------------------------------------------------------------------------
__global__ void lim_kernel(const float* __restrict__ x, float* __restrict__ lim,
                           int rows, int in_dim) {
    int m = blockIdx.x * blockDim.x + threadIdx.x;
    if (m >= rows) return;
    float x0 = x[(size_t)m * in_dim + 0];
    float x1 = x[(size_t)m * in_dim + 1];
    float r = 0.0f;
    if (x1 > 0.0f) {
        float a = x0 - 0.5f * x1;  a = a > 0.0f ? a : 0.0f;
        float b = 2.0f * x1 - x0;  b = b > 0.0f ? b : 0.0f;
        float d = 0.5f * x1;
        r = (a * b) / (d * d);
    }
    lim[m] = r;
}

// ---------------------------------------------------------------------------
// Tiled WMMA GEMM: C[M,N] = act(A[M,K](bf16) @ B[N,K]^T(bf16) + bias[N])
//   MODE 0: act = tanh, store bf16 (inter-layer activation)
//   MODE 1: act = (v)^2 * lim[row], store f32 (final output)
// Block: 256 threads (8 waves, 4 along M x 2 along N); BM=BN=128, BK=64.
// Wave tile 32x64 = 2x4 WMMA 16x16 accumulators. Double-buffered LDS with
// async global->LDS copies (ASYNCcnt), one barrier per K stage.
// ---------------------------------------------------------------------------
#define BM 128
#define BN 128
#define BK 64
#define LDK (BK + 8)   // ushort stride, +16B pad: rows land on distinct banks

template <int MODE>
__global__ __launch_bounds__(256)
void gemm_wmma_kernel(const unsigned short* __restrict__ A,   // M x K bf16
                      const unsigned short* __restrict__ B,   // N x K bf16
                      const float* __restrict__ bias,         // N  f32
                      unsigned short* __restrict__ Obf,       // M x N bf16 (MODE 0)
                      float* __restrict__ Of,                 // M x N f32  (MODE 1)
                      const float* __restrict__ lim,          // M          (MODE 1)
                      int M, int N, int K) {
    __shared__ alignas(16) unsigned short As[2][BM * LDK];
    __shared__ alignas(16) unsigned short Bs[2][BN * LDK];

    const int tid   = threadIdx.x;
    const int lane  = tid & 31;
    const int wave  = tid >> 5;
    const int waveM = wave & 3;       // 0..3
    const int waveN = wave >> 2;      // 0..1
    const int l15   = lane & 15;
    const int lhalf = lane >> 4;      // 0 or 1
    const int blockN0 = blockIdx.x * BN;
    const int blockM0 = blockIdx.y * BM;

    v8f acc[2][4];
    #pragma unroll
    for (int mi = 0; mi < 2; ++mi)
        #pragma unroll
        for (int ni = 0; ni < 4; ++ni)
            #pragma unroll
            for (int e = 0; e < 8; ++e) acc[mi][ni][e] = 0.0f;

    // Stage one 128x64 A tile + 128x64 B tile into LDS buffer `buf`.
    // 2048 chunks of 16B over 256 threads = 8 chunks/thread.
    auto stage = [&](int buf, int k0) {
        #pragma unroll
        for (int i = 0; i < 4; ++i) {
            int idx = tid + i * 256;          // 0..1023
            int row = idx >> 3;
            int c8  = (idx & 7) * 8;          // ushort offset, 16B aligned
#if USE_ASYNC_LDS
            __builtin_amdgcn_global_load_async_to_lds_b128(
                (gas_i32x4*)(A + (size_t)(blockM0 + row) * K + k0 + c8),
                (las_i32x4*)(&As[buf][row * LDK + c8]),
                0, 0);
            __builtin_amdgcn_global_load_async_to_lds_b128(
                (gas_i32x4*)(B + (size_t)(blockN0 + row) * K + k0 + c8),
                (las_i32x4*)(&Bs[buf][row * LDK + c8]),
                0, 0);
#else
            *reinterpret_cast<u32x4*>(&As[buf][row * LDK + c8]) =
                *reinterpret_cast<const u32x4*>(&A[(size_t)(blockM0 + row) * K + k0 + c8]);
            *reinterpret_cast<u32x4*>(&Bs[buf][row * LDK + c8]) =
                *reinterpret_cast<const u32x4*>(&B[(size_t)(blockN0 + row) * K + k0 + c8]);
#endif
        }
    };

    stage(0, 0);
    async_wait0();
    __syncthreads();

    int cur = 0;
    for (int k0 = 0; k0 < K; k0 += BK) {
        int nxt = cur ^ 1;
        if (k0 + BK < K) stage(nxt, k0 + BK);   // overlaps with WMMA below

        #pragma unroll
        for (int ks = 0; ks < BK; ks += 32) {
            // A frags: 16x32, lane(0-15)=M, lo-half K=0..7 & 16..23, hi-half +8
            AFrag a[2];
            #pragma unroll
            for (int mi = 0; mi < 2; ++mi) {
                int row  = waveM * 32 + mi * 16 + l15;
                int base = ks + lhalf * 8;
                a[mi].h[0] = *reinterpret_cast<const u16x8*>(&As[cur][row * LDK + base]);
                a[mi].h[1] = *reinterpret_cast<const u16x8*>(&As[cur][row * LDK + base + 16]);
            }
            // B frags: 32x16, lane(0-15)=N, lo-half K=0..7 & 16..23, hi-half +8
            AFrag b[4];
            #pragma unroll
            for (int ni = 0; ni < 4; ++ni) {
                int col  = waveN * 64 + ni * 16 + l15;
                int base = ks + lhalf * 8;
                b[ni].h[0] = *reinterpret_cast<const u16x8*>(&Bs[cur][col * LDK + base]);
                b[ni].h[1] = *reinterpret_cast<const u16x8*>(&Bs[cur][col * LDK + base + 16]);
            }
            #pragma unroll
            for (int mi = 0; mi < 2; ++mi)
                #pragma unroll
                for (int ni = 0; ni < 4; ++ni)
                    acc[mi][ni] = __builtin_amdgcn_wmma_f32_16x16x32_bf16(
                        false, a[mi].v, false, b[ni].v,
                        (short)0, acc[mi][ni], false, false);
        }

        async_wait0();       // my async writes into buf[nxt] have landed
        __syncthreads();     // everyone done reading buf[cur] / writing buf[nxt]
        cur = nxt;
    }

    // --- epilogue. D layout: VGPR e, lanes 0-15 -> M=e, lanes 16-31 -> M=e+8
    #pragma unroll
    for (int mi = 0; mi < 2; ++mi) {
        #pragma unroll
        for (int ni = 0; ni < 4; ++ni) {
            int col  = blockN0 + waveN * 64 + ni * 16 + l15;
            float bv = bias[col];
            #pragma unroll
            for (int e = 0; e < 8; ++e) {
                int row = blockM0 + waveM * 32 + mi * 16 + e + lhalf * 8;
                float v = acc[mi][ni][e] + bv;
                if (MODE == 0) {
                    Obf[(size_t)row * N + col] = f32_to_bf16_rne(DEV_TANH(v));
                } else {
                    Of[(size_t)row * N + col] = v * v * lim[row];
                }
            }
        }
    }
}

// ---------------------------------------------------------------------------
// Host: x,W1,b1,W2,b2,W3,b3,W4,b4 ; dims 16384x1024 -> 2048 -> 4096 -> 2048 -> 1024
// ---------------------------------------------------------------------------
extern "C" void kernel_launch(void* const* d_in, const int* in_sizes, int n_in,
                              void* d_out, int out_size, void* d_ws, size_t ws_size,
                              hipStream_t stream) {
    const int Mb = 16384, D0 = 1024, D1 = 2048, D2 = 4096, D3 = 2048, D4 = 1024;

    const float* x  = (const float*)d_in[0];
    const float* W1 = (const float*)d_in[1];
    const float* b1 = (const float*)d_in[2];
    const float* W2 = (const float*)d_in[3];
    const float* b2 = (const float*)d_in[4];
    const float* W3 = (const float*)d_in[5];
    const float* b3 = (const float*)d_in[6];
    const float* W4 = (const float*)d_in[7];
    const float* b4 = (const float*)d_in[8];

    char* ws = (char*)d_ws;
    size_t off = 0;
    auto alloc = [&](size_t bytes) -> void* {
        void* p = ws + off;
        off += (bytes + 255) & ~(size_t)255;
        return p;
    };
    unsigned short* xb  = (unsigned short*)alloc((size_t)Mb * D0 * 2);   //  32 MB
    unsigned short* w1b = (unsigned short*)alloc((size_t)D1 * D0 * 2);   //   4 MB
    unsigned short* w2b = (unsigned short*)alloc((size_t)D2 * D1 * 2);   //  16 MB
    unsigned short* w3b = (unsigned short*)alloc((size_t)D3 * D2 * 2);   //  16 MB
    unsigned short* w4b = (unsigned short*)alloc((size_t)D4 * D3 * 2);   //   4 MB
    unsigned short* h1  = (unsigned short*)alloc((size_t)Mb * D1 * 2);   //  64 MB
    unsigned short* h2  = (unsigned short*)alloc((size_t)Mb * D2 * 2);   // 128 MB
    unsigned short* h3  = (unsigned short*)alloc((size_t)Mb * D3 * 2);   //  64 MB
    float*          lb  = (float*)alloc((size_t)Mb * 4);                 //  64 KB

    auto cvt = [&](const float* in, unsigned short* out, size_t n) {
        int n4 = (int)(n / 4);
        int blocks = (n4 + 255) / 256;
        if (blocks > 4096) blocks = 4096;
        cvt_bf16_kernel<<<blocks, 256, 0, stream>>>((const float4*)in, (uint2*)out, n4);
    };
    cvt(x,  xb,  (size_t)Mb * D0);
    cvt(W1, w1b, (size_t)D1 * D0);
    cvt(W2, w2b, (size_t)D2 * D1);
    cvt(W3, w3b, (size_t)D3 * D2);
    cvt(W4, w4b, (size_t)D4 * D3);

    lim_kernel<<<(Mb + 255) / 256, 256, 0, stream>>>(x, lb, Mb, D0);

    dim3 blk(256);
    gemm_wmma_kernel<0><<<dim3(D1 / BN, Mb / BM), blk, 0, stream>>>(
        xb, w1b, b1, h1, nullptr, nullptr, Mb, D1, D0);
    gemm_wmma_kernel<0><<<dim3(D2 / BN, Mb / BM), blk, 0, stream>>>(
        h1, w2b, b2, h2, nullptr, nullptr, Mb, D2, D1);
    gemm_wmma_kernel<0><<<dim3(D3 / BN, Mb / BM), blk, 0, stream>>>(
        h2, w3b, b3, h3, nullptr, nullptr, Mb, D3, D2);
    gemm_wmma_kernel<1><<<dim3(D4 / BN, Mb / BM), blk, 0, stream>>>(
        h3, w4b, b4, nullptr, (float*)d_out, lb, Mb, D4, D3);
}
